// Model_26096221290674
// MI455X (gfx1250) — matrix operand order repaired and checked by reading, not compiled, not run on hardware
//
#include <hip/hip_runtime.h>
#include <math.h>

// ---------------------------------------------------------------------------
// CDNA5 WMMA (wave32): A = 16x32 bf16 (K-pairs per lane),
// B = 32x16 bf16 (lane = N column, 16 K values per lane), C/D = 16x16 f32.
// ---------------------------------------------------------------------------
typedef __attribute__((ext_vector_type(16))) __bf16 v16bf;
typedef __attribute__((ext_vector_type(8)))  float  v8f;
typedef int v4i __attribute__((vector_size(16)));

#define GLOBAL_AS __attribute__((address_space(1)))
#define LDS_AS    __attribute__((address_space(3)))

#if defined(__has_builtin)
#if __has_builtin(__builtin_amdgcn_global_load_async_to_lds_b128) && \
    __has_builtin(__builtin_amdgcn_s_wait_asynccnt)
#define USE_ASYNC_LDS 1
#endif
#endif

union FragBF {
  uint4 q[2];   // 32 bytes
  v16bf v;
};

template <typename T>
__device__ __forceinline__ T loadg(const T* p) {
  return *(const GLOBAL_AS T*)(const GLOBAL_AS void*)p;
}
template <typename T>
__device__ __forceinline__ void storeg(T* p, T v) {
  *(GLOBAL_AS T*)(GLOBAL_AS void*)p = v;
}

__device__ __forceinline__ unsigned short f2bf(float f) {
  unsigned u = __float_as_uint(f);
  unsigned r = u + 0x7FFFu + ((u >> 16) & 1u);   // round-to-nearest-even
  return (unsigned short)(r >> 16);
}

__device__ __forceinline__ float leaky_tanh_f(float x) {
  return tanhf(x * (2.0f / 3.0f)) * 1.7159f + 0.01f * x;
}

__device__ __forceinline__ float sigmoid_f(float x) {
  return 1.0f / (1.0f + expf(-x));
}

// 16B global -> LDS copy: async DMA (ASYNCcnt) if available, else sync.
// NOTE: use plain vector v4i here (not uint4/HIP_vector_type) so the host
// pass can type-check the fallback across address spaces.
__device__ __forceinline__ void async_copy16(const void* gsrc, unsigned* ldst) {
#if USE_ASYNC_LDS
  __builtin_amdgcn_global_load_async_to_lds_b128(
      (GLOBAL_AS v4i*)(void*)gsrc, (LDS_AS v4i*)(void*)ldst, 0, 0);
#else
  const v4i v = *(const GLOBAL_AS v4i*)(const GLOBAL_AS void*)gsrc;
  *(v4i*)ldst = v;
#endif
}

__device__ __forceinline__ void wait_async_keep_next() {
#if USE_ASYNC_LDS
  __builtin_amdgcn_s_wait_asynccnt(6);   // 6 async ops per thread per chunk
#endif
}
__device__ __forceinline__ void wait_async_all() {
#if USE_ASYNC_LDS
  __builtin_amdgcn_s_wait_asynccnt(0);
#endif
}

// ---------------------------------------------------------------------------
// conversions
// ---------------------------------------------------------------------------
__global__ void cvt_bf16_kernel(const float* __restrict__ s,
                                unsigned short* __restrict__ d, int n) {
  int i = blockIdx.x * 256 + threadIdx.x;
  if (i < n) storeg(d + i, f2bf(loadg(s + i)));
}

// inputs [T][B][I] fp32 -> [T][I][B] bf16 (per-timestep transpose)
__global__ void cvt_inputs_T_kernel(const float* __restrict__ s,
                                    unsigned short* __restrict__ d,
                                    int TT, int Bm, int Ii) {
  int idx = blockIdx.x * 256 + threadIdx.x;
  if (idx >= TT * Bm * Ii) return;
  int t = idx / (Bm * Ii);
  int r = idx - t * Bm * Ii;
  int m = r / Ii;
  int k = r - m * Ii;
  storeg(d + (size_t)t * Ii * Bm + (size_t)k * Bm + m, f2bf(loadg(s + idx)));
}

__global__ void zero_u32_kernel(unsigned* __restrict__ p, int n) {
  int i = blockIdx.x * 256 + threadIdx.x;
  if (i < n) storeg(p + i, 0u);
}

// ---------------------------------------------------------------------------
// GEMM:  D[64 x N] = act( X1[64 x K1] W1[N x K1]^T + X2[64 x K2] W2[N x K2]^T
//                         + b1 + b2 )
// Activations are given TRANSPOSED: XT[K][64] bf16 (== WMMA B layout).
// Weights W row-major [N][K] bf16 (== WMMA A layout per n-row).
// outF : fp32 [64][N] (row-major, m-major)    (nullable)
// outBT: bf16 [N][64] (transposed, feeds the next GEMM as XT)  (nullable)
// act: 0 = none, 1 = leaky_tanh
// Block: 256 threads (8 waves), tile = 128 n-rows x 64 m-cols.
// Wave w owns n-rows [16w,16w+16), all 64 m => 4 accumulators.
// K chunks of 64 double-buffered through LDS via async global->LDS DMA.
// ---------------------------------------------------------------------------
#define LSTR 36   // 64 bf16 = 32 uints, padded to 36 (144B, 16B aligned)

__global__ __launch_bounds__(256)
void gemm_bf16_kernel(const unsigned short* __restrict__ XT1,
                      const unsigned short* __restrict__ W1, int K1,
                      const unsigned short* __restrict__ XT2,
                      const unsigned short* __restrict__ W2, int K2,
                      const float* __restrict__ bias1,
                      const float* __restrict__ bias2,
                      int N, int act,
                      float* __restrict__ outF,
                      unsigned short* __restrict__ outBT) {
  __shared__ unsigned lds_w[2][128 * LSTR];  // W tile  [n_local][k-pairs]
  __shared__ unsigned lds_x[2][64 * LSTR];   // XT tile [k_local][m-pairs]

  const int tid  = threadIdx.x;
  const int wave = tid >> 5;
  const int lane = tid & 31;
  const int r    = lane & 15;
  const int kh   = lane >> 4;
  const int nBase = blockIdx.x * 128;

  v8f acc[4];
#pragma unroll
  for (int j = 0; j < 4; ++j)
    acc[j] = (v8f){0.f, 0.f, 0.f, 0.f, 0.f, 0.f, 0.f, 0.f};

  const int nch1 = K1 >> 6;
  const int nch2 = K2 >> 6;
  const int nch  = nch1 + nch2;

  auto issue = [&](int c, int buf) {
    const unsigned short* W  = (c < nch1) ? W1 : W2;
    const unsigned short* XT = (c < nch1) ? XT1 : XT2;
    const int K  = (c < nch1) ? K1 : K2;
    const int kb = ((c < nch1) ? c : (c - nch1)) << 6;
    unsigned* lw = lds_w[buf];
    unsigned* lx = lds_x[buf];
    // W tile: 128 rows x 64 bf16 = 1024 x 16B chunks, 4 per thread
#pragma unroll
    for (int it = 0; it < 4; ++it) {
      const int idx = tid + it * 256;
      const int row = idx >> 3;            // 0..127
      const int g   = idx & 7;             // 16B group
      int rowg = nBase + row;
      if (rowg >= N) rowg = N - 1;         // clamp (keeps loads in-bounds)
      const unsigned short* src = W + (size_t)rowg * K + kb + g * 8;
      async_copy16(src, lw + row * LSTR + g * 4);
      if (it == 0 && kb + 128 < K)
        __builtin_prefetch(src + 128, 0, 1);
    }
    // XT tile: 64 rows x 64 bf16 = 512 x 16B chunks, 2 per thread
#pragma unroll
    for (int it = 0; it < 2; ++it) {
      const int idx = tid + it * 256;
      const int row = idx >> 3;            // 0..63
      const int g   = idx & 7;
      const unsigned short* src = XT + (size_t)(kb + row) * 64 + g * 8;
      async_copy16(src, lx + row * LSTR + g * 4);
    }
  };

  issue(0, 0);
  for (int c = 0; c < nch; ++c) {
    const int buf = c & 1;
    if (c + 1 < nch) {
      issue(c + 1, buf ^ 1);
      wait_async_keep_next();
    } else {
      wait_async_all();
    }
    __syncthreads();

    const unsigned* lw = lds_w[buf];
    const unsigned* lx = lds_x[buf];
#pragma unroll
    for (int kc = 0; kc < 64; kc += 32) {
      FragBF a;
      const unsigned* ar = lw + (wave * 16 + r) * LSTR + (kc >> 1);
      a.q[0] = *(const uint4*)(ar + 4 * kh);
      a.q[1] = *(const uint4*)(ar + 8 + 4 * kh);
      const unsigned short* bx = (const unsigned short*)lx;
#pragma unroll
      for (int j = 0; j < 4; ++j) {
        FragBF b;
#pragma unroll
        for (int s = 0; s < 16; ++s) {
          const int k = kc + ((s < 8) ? (8 * kh + s) : (16 + 8 * kh + (s - 8)));
          b.v[s] = __builtin_bit_cast(__bf16, bx[k * (2 * LSTR) + j * 16 + r]);
        }
        acc[j] = __builtin_amdgcn_wmma_f32_16x16x32_bf16(
            false, a.v, false, b.v, (short)0, acc[j], false, false);
      }
    }
    __syncthreads();
  }

  // Epilogue. C/D layout: lane<16 -> M=vr (A row = n), N=lane (= m);
  //                       lane>=16 -> M=8+vr.
  const int lm = lane & 15;
#pragma unroll
  for (int vr = 0; vr < 8; ++vr) {
    const int ng  = nBase + wave * 16 + kh * 8 + vr;
    const bool nok = (ng < N);
    float bAdd = 0.f;
    if (nok) {
      if (bias1) bAdd = loadg(bias1 + ng);
      if (bias2) bAdd += loadg(bias2 + ng);
    }
#pragma unroll
    for (int j = 0; j < 4; ++j) {
      const int m = j * 16 + lm;
      float v = acc[j][vr] + bAdd;
      if (act == 1) v = leaky_tanh_f(v);
      if (nok) {
        if (outF)  storeg(outF + (size_t)m * N + ng, v);
        if (outBT) storeg(outBT + (size_t)ng * 64 + m, f2bf(v));
      }
    }
  }
}

// ---------------------------------------------------------------------------
// LSTM cell elementwise: gates G[64 x 4H] (i,f,g,o), c fp32 in/out,
// h written TRANSPOSED bf16: hT[H][64].
// ---------------------------------------------------------------------------
__global__ void lstm_cell_kernel(const float* __restrict__ G,
                                 float* __restrict__ c,
                                 unsigned short* __restrict__ hT, int Hc) {
  const int idx = blockIdx.x * 256 + threadIdx.x;   // over 64*Hc
  const int m = idx / Hc;
  const int j = idx - m * Hc;
  const float* g = G + (size_t)m * 4 * Hc;
  const float iv = sigmoid_f(loadg(g + j));
  const float fv = sigmoid_f(loadg(g + Hc + j));
  const float gv = tanhf(loadg(g + 2 * Hc + j));
  const float ov = sigmoid_f(loadg(g + 3 * Hc + j));
  const float cn = fv * loadg(c + idx) + iv * gv;
  storeg(c + idx, cn);
  storeg(hT + (size_t)j * 64 + m, f2bf(ov * tanhf(cn)));
}

// ---------------------------------------------------------------------------
// Host launcher
// ---------------------------------------------------------------------------
extern "C" void kernel_launch(void* const* d_in, const int* in_sizes, int n_in,
                              void* d_out, int out_size, void* d_ws, size_t ws_size,
                              hipStream_t stream) {
  enum { T = 512, B = 64, I = 128, H = 2048, H1 = 1024, H2 = 256, F2 = 512, O = 64 };

  const float* inputs = (const float*)d_in[0];
  const float* fc1_w  = (const float*)d_in[1];
  const float* fc1_b  = (const float*)d_in[2];
  const float* w_ih1  = (const float*)d_in[3];
  const float* w_hh1  = (const float*)d_in[4];
  const float* b_ih1  = (const float*)d_in[5];
  const float* b_hh1  = (const float*)d_in[6];
  const float* fc2_w  = (const float*)d_in[7];
  const float* fc2_b  = (const float*)d_in[8];
  const float* w_ih2  = (const float*)d_in[9];
  const float* w_hh2  = (const float*)d_in[10];
  const float* b_ih2  = (const float*)d_in[11];
  const float* b_hh2  = (const float*)d_in[12];
  const float* fc3_w  = (const float*)d_in[13];
  const float* fc3_b  = (const float*)d_in[14];
  float* out = (float*)d_out;

  // --- workspace carve-up (256B aligned) ---
  char* base = (char*)d_ws;
  size_t off = 0;
  auto alloc = [&](size_t bytes) -> void* {
    void* p = base + off;
    off += (bytes + 255) & ~(size_t)255;
    return p;
  };
  unsigned short* inT_bf  = (unsigned short*)alloc((size_t)T * I * B * 2);  // [T][I][B]
  unsigned short* fc1w_bf = (unsigned short*)alloc((size_t)H * I * 2);
  unsigned short* wih1_bf = (unsigned short*)alloc((size_t)4 * H1 * H * 2);
  unsigned short* whh1_bf = (unsigned short*)alloc((size_t)4 * H1 * H1 * 2);
  unsigned short* fc2w_bf = (unsigned short*)alloc((size_t)F2 * H1 * 2);
  unsigned short* wih2_bf = (unsigned short*)alloc((size_t)4 * H2 * F2 * 2);
  unsigned short* whh2_bf = (unsigned short*)alloc((size_t)4 * H2 * H2 * 2);
  unsigned short* fc3w_bf = (unsigned short*)alloc((size_t)O * H2 * 2);
  unsigned short* X1T = (unsigned short*)alloc((size_t)H * B * 2);    // [2048][64]
  unsigned short* X2T = (unsigned short*)alloc((size_t)F2 * B * 2);   // [512][64]
  float* G1 = (float*)alloc((size_t)B * 4 * H1 * 4);                  // [64][4096]
  float* G2 = (float*)alloc((size_t)B * 4 * H2 * 4);                  // [64][1024]
  unsigned short* h1T = (unsigned short*)alloc((size_t)H1 * B * 2);   // [1024][64]
  float*          c1  = (float*)alloc((size_t)B * H1 * 4);
  unsigned short* h2T = (unsigned short*)alloc((size_t)H2 * B * 2);   // [256][64]
  float*          c2  = (float*)alloc((size_t)B * H2 * 4);

  auto cvt = [&](const float* s, unsigned short* d, size_t n) {
    cvt_bf16_kernel<<<(unsigned)((n + 255) / 256), 256, 0, stream>>>(s, d, (int)n);
  };
  auto zero = [&](void* p, size_t n_u32) {
    zero_u32_kernel<<<(unsigned)((n_u32 + 255) / 256), 256, 0, stream>>>((unsigned*)p, (int)n_u32);
  };
  auto gemm = [&](const unsigned short* XT1, const unsigned short* W1, int K1,
                  const unsigned short* XT2, const unsigned short* W2, int K2,
                  const float* b1, const float* b2, int N, int act,
                  float* oF, unsigned short* oBT) {
    gemm_bf16_kernel<<<(N + 127) / 128, 256, 0, stream>>>(XT1, W1, K1, XT2, W2, K2,
                                                          b1, b2, N, act, oF, oBT);
  };

  // --- one-time (per call, deterministic) conversions + state init ---
  {
    size_t n = (size_t)T * B * I;
    cvt_inputs_T_kernel<<<(unsigned)((n + 255) / 256), 256, 0, stream>>>(
        inputs, inT_bf, T, B, I);
  }
  cvt(fc1_w, fc1w_bf, (size_t)H * I);
  cvt(w_ih1, wih1_bf, (size_t)4 * H1 * H);
  cvt(w_hh1, whh1_bf, (size_t)4 * H1 * H1);
  cvt(fc2_w, fc2w_bf, (size_t)F2 * H1);
  cvt(w_ih2, wih2_bf, (size_t)4 * H2 * F2);
  cvt(w_hh2, whh2_bf, (size_t)4 * H2 * H2);
  cvt(fc3_w, fc3w_bf, (size_t)O * H2);
  zero(h1T, (size_t)B * H1 / 2);
  zero(c1, (size_t)B * H1);
  zero(h2T, (size_t)B * H2 / 2);
  zero(c2, (size_t)B * H2);

  // --- sequential scan over timesteps ---
  for (int t = 0; t < T; ++t) {
    const unsigned short* xtT = inT_bf + (size_t)t * I * B;   // [128][64]

    // x = leaky_tanh(xt @ fc1_w^T + fc1_b)          -> X1T [2048][64] bf16
    gemm(xtT, fc1w_bf, I, nullptr, nullptr, 0, fc1_b, nullptr, H, 1, nullptr, X1T);

    // gates1 = X1 w_ih1^T + h1 w_hh1^T + biases     -> G1 [64][4096] f32
    gemm(X1T, wih1_bf, H, h1T, whh1_bf, H1, b_ih1, b_hh1, 4 * H1, 0, G1, nullptr);
    lstm_cell_kernel<<<(B * H1) / 256, 256, 0, stream>>>(G1, c1, h1T, H1);

    // x2 = leaky_tanh(h1 @ fc2_w^T + fc2_b)         -> X2T [512][64] bf16
    gemm(h1T, fc2w_bf, H1, nullptr, nullptr, 0, fc2_b, nullptr, F2, 1, nullptr, X2T);

    // gates2 = X2 w_ih2^T + h2 w_hh2^T + biases     -> G2 [64][1024] f32
    gemm(X2T, wih2_bf, F2, h2T, whh2_bf, H2, b_ih2, b_hh2, 4 * H2, 0, G2, nullptr);
    lstm_cell_kernel<<<(B * H2) / 256, 256, 0, stream>>>(G2, c2, h2T, H2);

    // yhat = h2 @ fc3_w^T + fc3_b                   -> d_out[t] [64][64] f32
    gemm(h2T, fc3w_bf, H2, nullptr, nullptr, 0, fc3_b, nullptr, O, 0,
         out + (size_t)t * B * O, nullptr);
  }
}
